// PointerGenerator_74766790689196
// MI455X (gfx1250) — compile-verified, matrix operand order checked by reading, not run on hardware
//
#include <hip/hip_runtime.h>

#define B_   16
#define SD_  64
#define SE_  400
#define D_   512
#define H_   8
#define V_   50000
#define EX_  200
#define ROWS (B_ * SD_)        // 1024
#define LDO  (V_ + EX_)        // 50200
#define VPAD 50048             // 782 * 64, padded vocab for full WMMA tiles

typedef _Float16 v8h  __attribute__((ext_vector_type(8)));
typedef _Float16 v16h __attribute__((ext_vector_type(16)));
typedef float    v8f  __attribute__((ext_vector_type(8)));

// ---------------- helpers ----------------

__device__ inline float wave_sum(float v) {
#pragma unroll
    for (int off = 16; off > 0; off >>= 1) v += __shfl_down(v, off, 32);
    return v;
}

// Build a v16h A/B fragment per 16-bit WMMA lane layout:
// elements 0..7  = halfs at p[0..7]   (K = k0 + hi*8 .. +7)
// elements 8..15 = halfs at p[16..23] (K = k0 + 16 + hi*8 .. +7)
__device__ inline v16h load_frag(const _Float16* __restrict__ p) {
    v8h lo = *(const v8h*)(p);
    v8h hh = *(const v8h*)(p + 16);
    v16h r;
#pragma unroll
    for (int i = 0; i < 8; ++i) { r[i] = lo[i]; r[i + 8] = hh[i]; }
    return r;
}

// ---------------- prep kernels ----------------

// W_vocab [D, V] fp32 -> Wt [VPAD, D] f16 (transposed, pad rows zeroed)
__global__ void PG_cvt_wt(const float* __restrict__ W, _Float16* __restrict__ Wt) {
    size_t i = (size_t)blockIdx.x * blockDim.x + threadIdx.x;
    if (i >= (size_t)VPAD * D_) return;
    int v = (int)(i >> 9);        // / 512
    int d = (int)(i & 511);
    _Float16 o = (_Float16)0.f;
    if (v < V_) o = (_Float16)W[(size_t)d * V_ + v];
    Wt[i] = o;
}

// decoder_outputs [1024, 512] fp32 -> f16 (same layout)
__global__ void PG_cvt_dec(const float* __restrict__ X, _Float16* __restrict__ Y) {
    int i = blockIdx.x * blockDim.x + threadIdx.x;
    if (i < ROWS * D_) Y[i] = (_Float16)X[i];
}

// copy_prob[b,q,s] = mean over heads of attn[b,h,q,s]
__global__ void PG_copyprob(const float* __restrict__ attn, float* __restrict__ cp) {
    int i = blockIdx.x * blockDim.x + threadIdx.x;
    if (i >= ROWS * SE_) return;
    int s = i % SE_;
    int q = (i / SE_) % SD_;
    int b = i / (SD_ * SE_);
    float acc = 0.f;
#pragma unroll
    for (int h = 0; h < H_; ++h)
        acc += attn[(((size_t)(b * H_ + h) * SD_) + q) * SE_ + s];
    cp[i] = acc * (1.0f / H_);
}

// ew[b,s] = enc[b,s,:] . w_h   (one wave per output)
__global__ void PG_ew(const float* __restrict__ enc, const float* __restrict__ wh,
                      float* __restrict__ ew) {
    int wid  = blockIdx.x * (blockDim.x / 32) + (threadIdx.x >> 5);
    int lane = threadIdx.x & 31;
    if (wid >= B_ * SE_) return;
    const float* e = enc + (size_t)wid * D_;
    float acc = 0.f;
    for (int d = lane; d < D_; d += 32) acc += e[d] * wh[d];
    acc = wave_sum(acc);
    if (lane == 0) ew[wid] = acc;
}

// p_gen[row] = sigmoid( cp[row,:].ew[b,:] + dec[row,:].w_s + tgt[row,:].w_x + b_h+b_s+b_x )
__global__ void PG_pgen(const float* __restrict__ cp, const float* __restrict__ ew,
                        const float* __restrict__ dec, const float* __restrict__ ws_,
                        const float* __restrict__ tgt, const float* __restrict__ wx,
                        const float* __restrict__ bh, const float* __restrict__ bs,
                        const float* __restrict__ bx,
                        float* __restrict__ pgen, float* __restrict__ pcopy) {
    int row  = blockIdx.x * (blockDim.x / 32) + (threadIdx.x >> 5);
    int lane = threadIdx.x & 31;
    if (row >= ROWS) return;
    int b = row / SD_;
    float acc = 0.f;
    const float* c = cp + (size_t)row * SE_;
    const float* e = ew + (size_t)b * SE_;
    for (int s = lane; s < SE_; s += 32) acc += c[s] * e[s];
    const float* dr = dec + (size_t)row * D_;
    const float* tr = tgt + (size_t)row * D_;
    for (int d = lane; d < D_; d += 32) acc += dr[d] * ws_[d] + tr[d] * wx[d];
    acc = wave_sum(acc);
    if (lane == 0) {
        float x = acc + bh[0] + bs[0] + bx[0];
        float g = 1.0f / (1.0f + __expf(-x));
        pgen[row]  = g;
        pcopy[row] = 1.0f - g;
    }
}

// ---------------- WMMA GEMM: logits = dec16 @ Wt^T (+ b_vocab) ----------------
// One wave computes a 32 (rows) x 64 (vocab cols) tile, K = 512 in steps of 32.
// Each B fragment feeds two WMMAs (M-reuse) -> ~1.5 b128 loads per WMMA.
__global__ __launch_bounds__(32) void PG_gemm(const _Float16* __restrict__ A,   // [1024,512]
                                              const _Float16* __restrict__ Bt,  // [VPAD,512]
                                              const float* __restrict__ bias,   // [V]
                                              float* __restrict__ out) {        // [1024,LDO]
    const int lane = threadIdx.x;
    const int m0   = blockIdx.y * 32;
    const int n0   = blockIdx.x * 64;
    const int l16  = lane & 15;
    const int hi   = lane >> 4;           // lanes 0-15 -> K lo groups, 16-31 -> K hi groups

    const _Float16* arow0 = A + (size_t)(m0 + l16) * D_ + hi * 8;
    const _Float16* arow1 = arow0 + (size_t)16 * D_;
    const _Float16* brow  = Bt + (size_t)(n0 + l16) * D_ + hi * 8;

    v8f acc[2][4];
#pragma unroll
    for (int a = 0; a < 2; ++a)
#pragma unroll
        for (int t = 0; t < 4; ++t) acc[a][t] = {};

    for (int k0 = 0; k0 < D_; k0 += 32) {
        v16h a0 = load_frag(arow0 + k0);
        v16h a1 = load_frag(arow1 + k0);
#pragma unroll
        for (int t = 0; t < 4; ++t) {
            v16h bfr = load_frag(brow + (size_t)(16 * t) * D_ + k0);
            acc[0][t] = __builtin_amdgcn_wmma_f32_16x16x32_f16(
                false, a0, false, bfr, (short)0, acc[0][t], false, false);
            acc[1][t] = __builtin_amdgcn_wmma_f32_16x16x32_f16(
                false, a1, false, bfr, (short)0, acc[1][t], false, false);
        }
    }

    // C/D layout: lane holds col N = l16, rows M = hi*8 + r in acc element r
#pragma unroll
    for (int t = 0; t < 4; ++t) {
        int n = n0 + 16 * t + l16;
        if (n < V_) {
            float bv = bias[n];
#pragma unroll
            for (int a = 0; a < 2; ++a) {
                size_t rbase = (size_t)(m0 + 16 * a + hi * 8);
#pragma unroll
                for (int r = 0; r < 8; ++r)
                    out[(rbase + r) * LDO + n] = acc[a][t][r] + bv;
            }
        }
    }
}

// ---------------- fused per-row softmax: max, sum, normalize, scale, pad ----------------
// One block per output row. Three strided passes over the (cache-hot) 200KB row.
__global__ __launch_bounds__(256) void PG_rowsoftmax(float* __restrict__ out,
                                                     const float* __restrict__ pgen) {
    __shared__ float red[256];
    int row = blockIdx.x, tid = threadIdx.x;
    float* p = out + (size_t)row * LDO;

    float mx = -3.4e38f;
    for (int c = tid; c < V_; c += 256) mx = fmaxf(mx, p[c]);
    red[tid] = mx; __syncthreads();
#pragma unroll
    for (int s = 128; s > 0; s >>= 1) {
        if (tid < s) red[tid] = fmaxf(red[tid], red[tid + s]);
        __syncthreads();
    }
    float m = red[0]; __syncthreads();

    float sm = 0.f;
    for (int c = tid; c < V_; c += 256) sm += __expf(p[c] - m);
    red[tid] = sm; __syncthreads();
#pragma unroll
    for (int s = 128; s > 0; s >>= 1) {
        if (tid < s) red[tid] += red[tid + s];
        __syncthreads();
    }
    float scale = pgen[row] / red[0];

    for (int c = tid; c < V_; c += 256) p[c] = __expf(p[c] - m) * scale;
    for (int c = V_ + tid; c < LDO; c += 256) p[c] = 0.f;
}

// ---------------- scatter-add of copy distribution, then log ----------------

__global__ void PG_scatter(const float* __restrict__ cp, const float* __restrict__ pcopy,
                           const int* __restrict__ widx, float* __restrict__ out) {
    int i = blockIdx.x * blockDim.x + threadIdx.x;
    if (i >= B_ * SD_ * SE_) return;
    int s = i % SE_;
    int t = (i / SE_) % SD_;
    int b = i / (SD_ * SE_);
    int row = b * SD_ + t;
    float add = cp[(size_t)row * SE_ + s] * pcopy[row];
    int col = widx[b * SE_ + s];
    atomicAdd(out + (size_t)row * LDO + col, add);
}

__global__ void PG_log(float* __restrict__ out) {
    size_t i = (size_t)blockIdx.x * blockDim.x + threadIdx.x;
    if (i >= (size_t)ROWS * LDO) return;
    out[i] = __logf(out[i] + 1e-9f);
}

// ---------------- launcher ----------------

extern "C" void kernel_launch(void* const* d_in, const int* in_sizes, int n_in,
                              void* d_out, int out_size, void* d_ws, size_t ws_size,
                              hipStream_t stream) {
    const float* enc  = (const float*)d_in[0];
    const float* dec  = (const float*)d_in[1];
    const float* attn = (const float*)d_in[2];
    const float* tgt  = (const float*)d_in[3];
    const int*   widx = (const int*)d_in[4];
    // d_in[5] extra_vocab (compile-time EX_), d_in[6] semantic_mask: unused (no-op in ref)
    const float* Wv   = (const float*)d_in[7];
    const float* bv   = (const float*)d_in[8];
    const float* wh   = (const float*)d_in[9];
    const float* bh   = (const float*)d_in[10];
    const float* w_s  = (const float*)d_in[11];
    const float* bs   = (const float*)d_in[12];
    const float* wx   = (const float*)d_in[13];
    const float* bx   = (const float*)d_in[14];
    float* out = (float*)d_out;

    char* p = (char*)d_ws;
    _Float16* Wt    = (_Float16*)p; p += (size_t)VPAD * D_ * sizeof(_Float16);
    _Float16* dec16 = (_Float16*)p; p += (size_t)ROWS * D_ * sizeof(_Float16);
    float* cp    = (float*)p; p += (size_t)ROWS * SE_ * sizeof(float);
    float* ew    = (float*)p; p += (size_t)B_ * SE_ * sizeof(float);
    float* pgen  = (float*)p; p += ROWS * sizeof(float);
    float* pcopy = (float*)p; p += ROWS * sizeof(float);
    (void)in_sizes; (void)n_in; (void)out_size; (void)ws_size;

    size_t nwt = (size_t)VPAD * D_;
    PG_cvt_wt <<<(unsigned)((nwt + 255) / 256), 256, 0, stream>>>(Wv, Wt);
    PG_cvt_dec<<<(ROWS * D_ + 255) / 256, 256, 0, stream>>>(dec, dec16);
    PG_copyprob<<<(ROWS * SE_ + 255) / 256, 256, 0, stream>>>(attn, cp);
    PG_ew  <<<(B_ * SE_ + 7) / 8, 256, 0, stream>>>(enc, wh, ew);
    PG_pgen<<<(ROWS + 7) / 8, 256, 0, stream>>>(cp, ew, dec, w_s, tgt, wx, bh, bs, bx,
                                                pgen, pcopy);

    dim3 ggrid(VPAD / 64, ROWS / 32);   // 782 x 32 waves, 32x64 tile each
    PG_gemm<<<ggrid, 32, 0, stream>>>(dec16, Wt, bv, out);

    PG_rowsoftmax<<<ROWS, 256, 0, stream>>>(out, pgen);

    PG_scatter<<<(B_ * SD_ * SE_ + 255) / 256, 256, 0, stream>>>(cp, pcopy, widx, out);

    size_t nout = (size_t)ROWS * LDO;
    PG_log<<<(unsigned)((nout + 255) / 256), 256, 0, stream>>>(out);
}
